// Tylor_expansion_PhiMLP_53961969107575
// MI455X (gfx1250) — compile-verified, matrix operand order checked by reading, not run on hardware
//
#include <hip/hip_runtime.h>
#include <hip/hip_bf16.h>
#include <math.h>

// ---------------------------------------------------------------------------
// Problem sizes (fixed by the reference): B*T = 8192, H = 2560, I = 10240
// ---------------------------------------------------------------------------
namespace cfg {
constexpr int MROWS = 8192;   // B*T
constexpr int HDIM  = 2560;   // hidden
constexpr int IPAD  = 10240;  // padded intermediate
constexpr int IREAL = 10236;  // real intermediate (I - S)
constexpr int SG    = 24;     // S*G
constexpr int BM = 128, BN = 128, BK = 64;
constexpr int LDST  = 72;               // LDS row stride (bf16), 144B = 9*16B
constexpr int STAGE = (BM + BN) * LDST; // bf16 elems per double-buffer stage
constexpr int LDS_BYTES = 2 * STAGE * 2;
}

typedef __attribute__((ext_vector_type(16))) __bf16 v16bf;
typedef __attribute__((ext_vector_type(8)))  __bf16 v8bf;
typedef __attribute__((ext_vector_type(8)))  float  v8f;
typedef __attribute__((ext_vector_type(4)))  int    v4i;

#if __has_builtin(__builtin_amdgcn_global_load_async_to_lds_b128)
#define HAS_ASYNC_BUILTIN 1
#else
#define HAS_ASYNC_BUILTIN 0
#endif

// Async copy of 16B global -> LDS (no VGPR data staging; tracked by ASYNCcnt).
// Builtin signature on this toolchain: (v4i AS1*, v4i AS3*, imm int, imm int).
__device__ __forceinline__ void async_copy16(const __bf16* gp, __bf16* lp) {
#if HAS_ASYNC_BUILTIN
  v4i* gpv = (v4i*)gp;   // generic bitcast (drops const)
  v4i* lpv = (v4i*)lp;
  __builtin_amdgcn_global_load_async_to_lds_b128(
      (__attribute__((address_space(1))) v4i*)gpv,
      (__attribute__((address_space(3))) v4i*)lpv,
      0, 0);
#else
  unsigned lofs = (unsigned)(size_t)(__attribute__((address_space(3))) void*)(lp);
  asm volatile("global_load_async_to_lds_b128 %0, %1, off"
               :: "v"(lofs), "v"(gp) : "memory");
#endif
}

__device__ __forceinline__ void wait_async_all() {
#if __has_builtin(__builtin_amdgcn_s_wait_asynccnt)
  __builtin_amdgcn_s_wait_asynccnt(0);
#else
  asm volatile("s_wait_asynccnt 0x0" ::: "memory");
#endif
}

// ---------------------------------------------------------------------------
// fp32 -> bf16 convert with zero padding (used for X, W1, W2)
// ---------------------------------------------------------------------------
__global__ void cvt_pad_bf16_kernel(const float* __restrict__ src,
                                    __bf16* __restrict__ dst,
                                    int srows, int scols, int drows, int dcols) {
  size_t total = (size_t)drows * dcols;
  size_t stride = (size_t)gridDim.x * blockDim.x;
  for (size_t i = (size_t)blockIdx.x * blockDim.x + threadIdx.x; i < total; i += stride) {
    int r = (int)(i / dcols);
    int c = (int)(i % dcols);
    float v = (r < srows && c < scols) ? src[(size_t)r * scols + c] : 0.0f;
    dst[i] = (__bf16)v;
  }
}

// ---------------------------------------------------------------------------
// Taylor-coefficient kernel: one block per row m.
// ---------------------------------------------------------------------------
__global__ __launch_bounds__(256)
void taylor_d_kernel(const float* __restrict__ X,
                     const float* __restrict__ w1s,
                     const float* __restrict__ lp,
                     float* __restrict__ dmat) {
  const int m = blockIdx.x;
  const int tid = threadIdx.x;
  const float* xr = X + (size_t)m * cfg::HDIM;

  float a0 = 0.f, a1 = 0.f, a2 = 0.f, a3 = 0.f;
  for (int k = tid; k < cfg::HDIM; k += 256) {
    float xv = xr[k];
    a0 += xv * w1s[0 * cfg::HDIM + k];
    a1 += xv * w1s[1 * cfg::HDIM + k];
    a2 += xv * w1s[2 * cfg::HDIM + k];
    a3 += xv * w1s[3 * cfg::HDIM + k];
  }
  __shared__ float red[4 * 256];
  red[0 * 256 + tid] = a0; red[1 * 256 + tid] = a1;
  red[2 * 256 + tid] = a2; red[3 * 256 + tid] = a3;
  __syncthreads();
  for (int ofs = 128; ofs > 0; ofs >>= 1) {
    if (tid < ofs) {
      red[0 * 256 + tid] += red[0 * 256 + tid + ofs];
      red[1 * 256 + tid] += red[1 * 256 + tid + ofs];
      red[2 * 256 + tid] += red[2 * 256 + tid + ofs];
      red[3 * 256 + tid] += red[3 * 256 + tid + ofs];
    }
    __syncthreads();
  }
  if (tid < 4) {
    float h1    = red[tid * 256];
    float delta = h1 - lp[tid];
    float sgn   = (delta > 0.f) ? 1.f : -1.f;
    float a     = fabsf(delta);
    bool  over  = (a > 2.0f);              // DELTA_THD
    float p = 1.f, fact = 1.f;
    float* dp = dmat + (size_t)m * cfg::SG + tid * 6;
    #pragma unroll
    for (int g = 0; g < 6; ++g) {
      p *= a;                              // a^(g+1)
      fact *= (float)(g + 1);              // (g+1)! == exp(discount_factor[g])
      float v = p / fact;
      if (((g + 1) & 1) == 1) v *= sgn;    // odd orders pick up the sign
      if ((g + 1) > 4 && over) v = 0.f;    // GRAD_ORDER_MIN truncation
      dp[g] = v;
    }
  }
}

// ---------------------------------------------------------------------------
// LDS fragment load matching the CDNA5 16-bit A/B WMMA layout:
//   lanes 0-15 hold K = 0..7 & 16..23, lanes 16-31 hold K = 8..15 & 24..31
// ---------------------------------------------------------------------------
__device__ __forceinline__ v16bf load_frag(const __bf16* lds, int row0, int lane, int kofs) {
  const int half = lane >> 4;
  const int rl   = lane & 15;
  const __bf16* base = lds + (row0 + rl) * cfg::LDST + kofs + half * 8;
  v8bf lo = *(const v8bf*)(base);
  v8bf hi = *(const v8bf*)(base + 16);
  v16bf v;
  #pragma unroll
  for (int i = 0; i < 8; ++i) { v[i] = lo[i]; v[i + 8] = hi[i]; }
  return v;
}

// ---------------------------------------------------------------------------
// NT bf16 GEMM: C[M,N] = A[M,K] * B[N,K]^T, fp32 accum, WMMA bf16.
// Block 128x128, BK=64, 8 waves 4(M) x 2(N); wave tile 32x64 (2x4 WMMAs).
// LDS double-buffered, filled by async global->LDS (ASYNCcnt tracked).
// EPI==0 : out_bf16 = gelu(acc + bias[n])
// EPI==1 : out_f32  = acc + lob[n] + d[m,:24].fw[n,:24]
// ---------------------------------------------------------------------------
template <int EPI>
__global__ __launch_bounds__(256)
void gemm_bf16_kernel(const __bf16* __restrict__ A, const __bf16* __restrict__ B,
                      int K,
                      const float* __restrict__ bias, int nreal,
                      const float* __restrict__ lob,
                      const float* __restrict__ dmat,
                      const float* __restrict__ fw,
                      __bf16* __restrict__ outbf, int ldo_bf,
                      float* __restrict__ outf, int ldo_f) {
  using namespace cfg;
  extern __shared__ __align__(16) char smem_raw[];
  __bf16* lds = (__bf16*)smem_raw;

  const int tid  = threadIdx.x;
  const int lane = tid & 31;
  const int wave = tid >> 5;
  const int wm   = wave & 3;   // 0..3  -> 32-row slice
  const int wn   = wave >> 2;  // 0..1  -> 64-col slice
  const int bM   = blockIdx.y * BM;
  const int bN   = blockIdx.x * BN;
  const int nk   = K / BK;

  // Issue this thread's share of one K-tile (4x16B of A, 4x16B of B) as
  // async global->LDS transfers into the given stage.
  auto issue_tile = [&](int kb, int st) {
    __bf16* dstA = lds + st * STAGE;
    __bf16* dstB = dstA + BM * LDST;
    #pragma unroll
    for (int i = 0; i < 4; ++i) {
      int c = tid + i * 256;
      int r = c >> 3;
      int col = (c & 7) * 8;
      async_copy16(A + (size_t)(bM + r) * K + (size_t)kb * BK + col,
                   dstA + r * LDST + col);
      async_copy16(B + (size_t)(bN + r) * K + (size_t)kb * BK + col,
                   dstB + r * LDST + col);
    }
  };

  v8f acc[2][4];
  for (int i = 0; i < 2; ++i)
    for (int j = 0; j < 4; ++j)
      for (int e = 0; e < 8; ++e) acc[i][j][e] = 0.f;

  issue_tile(0, 0);

  for (int kb = 0; kb < nk; ++kb) {
    const int st = kb & 1;
    wait_async_all();     // this wave's tile-kb transfers landed in LDS
    __syncthreads();      // all waves landed; previous compute done with st^1
    if (kb + 1 < nk) issue_tile(kb + 1, st ^ 1);  // overlap fill with compute

    const __bf16* ldsA = lds + st * STAGE;
    const __bf16* ldsB = ldsA + BM * LDST;
    #pragma unroll
    for (int ks = 0; ks < BK; ks += 32) {
      v16bf af[2], bfr[4];
      #pragma unroll
      for (int mt = 0; mt < 2; ++mt) af[mt] = load_frag(ldsA, wm * 32 + mt * 16, lane, ks);
      #pragma unroll
      for (int nt = 0; nt < 4; ++nt) bfr[nt] = load_frag(ldsB, wn * 64 + nt * 16, lane, ks);
      #pragma unroll
      for (int mt = 0; mt < 2; ++mt)
        #pragma unroll
        for (int nt = 0; nt < 4; ++nt)
          acc[mt][nt] = __builtin_amdgcn_wmma_f32_16x16x32_bf16(
              false, af[mt], false, bfr[nt], (short)0, acc[mt][nt], false, false);
    }
  }

  const int half = lane >> 4;
  const int nl   = lane & 15;

  if constexpr (EPI == 0) {
    // C layout: VGPR j -> M = j + 8*half, N = lane&15 (per 16x16 tile)
    #pragma unroll
    for (int mt = 0; mt < 2; ++mt)
      for (int nt = 0; nt < 4; ++nt)
        #pragma unroll
        for (int j = 0; j < 8; ++j) {
          int m = bM + wm * 32 + mt * 16 + j + half * 8;
          int n = bN + wn * 64 + nt * 16 + nl;
          float x = acc[mt][nt][j] + ((n < nreal) ? bias[n] : 0.f);
          float g = 0.5f * x * (1.f + erff(x * 0.70710678118654752f));
          outbf[(size_t)m * ldo_bf + n] = (__bf16)g;
        }
  } else {
    // Stage Taylor coefficients + fuse weights + local_out_bias for this tile.
    __syncthreads();
    float* sd   = (float*)lds;            // 128 rows x 24
    float* slob = sd + 128 * SG;          // 128
    float* sfw  = slob + 128;             // 128 cols x 24
    for (int i = tid; i < 128 * SG; i += 256) {
      int r = i / SG, c = i % SG;
      sd[i]  = dmat[(size_t)(bM + r) * SG + c];
      sfw[i] = fw[(size_t)(bN + r) * SG + c];
    }
    for (int i = tid; i < 128; i += 256) slob[i] = lob[bN + i];
    __syncthreads();
    #pragma unroll
    for (int mt = 0; mt < 2; ++mt)
      for (int nt = 0; nt < 4; ++nt)
        #pragma unroll
        for (int j = 0; j < 8; ++j) {
          int ml = wm * 32 + mt * 16 + j + half * 8;
          int nc = wn * 64 + nt * 16 + nl;
          float s = acc[mt][nt][j] + slob[nc];
          const float* dp = sd + ml * SG;
          const float* fp = sfw + nc * SG;
          #pragma unroll
          for (int t = 0; t < SG; ++t) s += dp[t] * fp[t];
          outf[(size_t)(bM + ml) * ldo_f + (bN + nc)] = s;
        }
  }
}

// ---------------------------------------------------------------------------
// Launch
// ---------------------------------------------------------------------------
extern "C" void kernel_launch(void* const* d_in, const int* in_sizes, int n_in,
                              void* d_out, int out_size, void* d_ws, size_t ws_size,
                              hipStream_t stream) {
  using namespace cfg;
  (void)in_sizes; (void)n_in; (void)out_size; (void)ws_size;

  const float* X   = (const float*)d_in[0];  // (8192, 2560)
  const float* w1s = (const float*)d_in[1];  // (4, 2560)
  const float* lp  = (const float*)d_in[2];  // (4,)
  const float* fw  = (const float*)d_in[3];  // (2560, 24)
  /* d_in[4] = discount_factor (recomputed as factorials on device) */
  const float* lob = (const float*)d_in[5];  // (2560,)
  const float* W1  = (const float*)d_in[6];  // (10236, 2560)
  const float* b1  = (const float*)d_in[7];  // (10236,)
  const float* W2  = (const float*)d_in[8];  // (2560, 10236)
  float* out = (float*)d_out;                // (8192, 2560)

  size_t off = 0;
  auto wsAlloc = [&](size_t bytes) -> char* {
    char* p = (char*)d_ws + off;
    off += (bytes + 255) & ~size_t(255);
    return p;
  };
  __bf16* Xbf   = (__bf16*)wsAlloc((size_t)MROWS * HDIM * 2);  // 42 MB
  __bf16* W1bf  = (__bf16*)wsAlloc((size_t)IPAD * HDIM * 2);   // 52 MB (row-padded)
  __bf16* W2bf  = (__bf16*)wsAlloc((size_t)HDIM * IPAD * 2);   // 52 MB (col-padded)
  __bf16* Actbf = (__bf16*)wsAlloc((size_t)MROWS * IPAD * 2);  // 168 MB
  float*  Dmat  = (float*)wsAlloc((size_t)MROWS * SG * 4);     // 0.8 MB

  // 1) fp32 -> bf16 converts (+ zero padding so GEMMs need no bounds checks)
  cvt_pad_bf16_kernel<<<2048, 256, 0, stream>>>(X,  Xbf,  MROWS, HDIM,  MROWS, HDIM);
  cvt_pad_bf16_kernel<<<2048, 256, 0, stream>>>(W1, W1bf, IREAL, HDIM,  IPAD,  HDIM);
  cvt_pad_bf16_kernel<<<2048, 256, 0, stream>>>(W2, W2bf, HDIM,  IREAL, HDIM,  IPAD);

  // 2) Taylor coefficients (fp32, tiny)
  taylor_d_kernel<<<MROWS, 256, 0, stream>>>(X, w1s, lp, Dmat);

  // 3) GEMM1: act = gelu(X @ W1^T + b1)   [8192 x 10240, K = 2560]
  dim3 g1(IPAD / BN, MROWS / BM);
  gemm_bf16_kernel<0><<<g1, 256, LDS_BYTES, stream>>>(Xbf, W1bf, HDIM,
                                                      b1, IREAL,
                                                      nullptr, nullptr, nullptr,
                                                      Actbf, IPAD, nullptr, 0);

  // 4) GEMM2: out = act @ W2^T + local_out_bias + einsum(d, fuse_w)
  dim3 g2(HDIM / BN, MROWS / BM);
  gemm_bf16_kernel<1><<<g2, 256, LDS_BYTES, stream>>>(Actbf, W2bf, IPAD,
                                                      nullptr, 0,
                                                      lob, Dmat, fw,
                                                      nullptr, 0, out, HDIM);
}
